// ContextAwareModel_56831007260907
// MI455X (gfx1250) — compile-verified, matrix operand order checked by reading, not run on hardware
//
#include <hip/hip_runtime.h>
#include <hip/hip_bf16.h>
#include <stdint.h>

// ---------------------------------------------------------------------------
// Model dims
// ---------------------------------------------------------------------------
#define Bq   32
#define Tq   128
#define Eq   768
#define Hq   256
#define G4H  1024            // 4*H
#define Mq   (Tq * Bq)       // 4096 rows in the big GEMMs
#define Dq   (Eq + 2 * Hq)   // 1280
#define HALFq (Dq / 2)       // 640

typedef __attribute__((ext_vector_type(8)))  float  v8f;
typedef __attribute__((ext_vector_type(16))) __bf16 v16bf;
typedef __attribute__((ext_vector_type(8)))  __bf16 v8bf;

__device__ __forceinline__ float sigmoidf(float x) {
    return 1.0f / (1.0f + __expf(-x));
}

// Build a 16x32 bf16 WMMA fragment for this lane from a row-major matrix.
// Lane l covers K offsets (l>>4)*8 .. +7  and  16+(l>>4)*8 .. +7  (two b128s).
__device__ __forceinline__ v16bf load_frag(const __bf16* rowk, int ksel) {
    const v8bf lo = *reinterpret_cast<const v8bf*>(rowk + ksel);
    const v8bf hi = *reinterpret_cast<const v8bf*>(rowk + 16 + ksel);
    v16bf r;
#pragma unroll
    for (int i = 0; i < 8; ++i) { r[i] = lo[i]; r[i + 8] = hi[i]; }
    return r;
}

#define WMMA_BF16(a, b, c) \
    __builtin_amdgcn_wmma_f32_16x16x32_bf16(false, (a), false, (b), (short)0, (c), false, false)

// Async-copy 16 bytes per lane from global to LDS (ASYNCcnt-tracked).
// LDS address = LDS base + the lane's VGPR offset (wave-relative byte offset = low 32 bits of
// the generic pointer for the LDS aperture, per ISA 10.2).
__device__ __forceinline__ void async_copy_b128(uint32_t lds_byte_off, const void* gaddr) {
    asm volatile("global_load_async_to_lds_b128 %0, %1, off"
                 :: "v"(lds_byte_off), "v"(gaddr)
                 : "memory");
}
__device__ __forceinline__ void wait_asynccnt0() {
    asm volatile("s_wait_asynccnt 0x0" ::: "memory");
}

// ---------------------------------------------------------------------------
// 1) Gather article embeddings -> bf16 A0 [4096, 768]   (row m = t*B + b)
// ---------------------------------------------------------------------------
__global__ void gather_embed(const int* __restrict__ article,
                             const float* __restrict__ emb,
                             __bf16* __restrict__ A0) {
    int i = blockIdx.x * blockDim.x + threadIdx.x;     // < 4096*768
    int e = i % Eq;
    int m = i / Eq;
    int t = m / Bq, b = m % Bq;
    int idx = article[b * Tq + t];
    A0[i] = (__bf16)emb[(size_t)idx * Eq + e];
}

// target_sent_reps -> written straight into d_out (+128 floats)
__global__ void target_reps(const int* __restrict__ article,
                            const int* __restrict__ positions,
                            const float* __restrict__ emb,
                            float* __restrict__ out) {
    int i = blockIdx.x * blockDim.x + threadIdx.x;     // < 32*768
    int b = i / Eq, e = i % Eq;
    int idx = article[b * Tq + positions[b]];
    out[i] = emb[(size_t)idx * Eq + e];
}

// f32 -> bf16 weight conversion (layout-preserving)
__global__ void cvt_bf16(const float* __restrict__ src, __bf16* __restrict__ dst, int n) {
    int i = blockIdx.x * blockDim.x + threadIdx.x;
    if (i < n) dst[i] = (__bf16)src[i];
}

// ---------------------------------------------------------------------------
// 2) Input-projection GEMM: out[dir][m][n] = sum_k A[m,k]*W[dir][n,k] + bias
//    A: bf16 [4096,K] row-major, W: bf16 [2][1024,K] row-major.
//    One wave per 32x32 block (2x2 WMMA tiles). K is a compile-time constant,
//    the K loop is fully unrolled: the scheduler hoists fragment loads across
//    WMMAs with no register rotation.
// ---------------------------------------------------------------------------
template<int K>
__global__ __launch_bounds__(256) void gemm_xproj(
        const __bf16* __restrict__ A, const __bf16* __restrict__ W,
        const float* __restrict__ bih, const float* __restrict__ bhh,
        float* __restrict__ out) {
    const int NT2 = G4H / 32;   // 32 col blocks
    const int MT2 = Mq  / 32;   // 128 row blocks
    int tid  = blockIdx.x * blockDim.x + threadIdx.x;
    int wave = tid >> 5;
    int lane = tid & 31;
    int dir  = wave / (MT2 * NT2);
    int rem  = wave % (MT2 * NT2);
    int mt = rem / NT2, nt = rem % NT2;
    int lr = lane & 15, ksel = (lane >> 4) * 8, rsel = (lane >> 4) * 8;

    const __bf16* ar0 = A + (size_t)(mt * 32 + lr) * K;
    const __bf16* ar1 = ar0 + (size_t)16 * K;
    const __bf16* wr0 = W + (size_t)dir * G4H * K + (size_t)(nt * 32 + lr) * K;
    const __bf16* wr1 = wr0 + (size_t)16 * K;

    v8f acc00 = {}, acc01 = {}, acc10 = {}, acc11 = {};
#pragma unroll
    for (int kk = 0; kk < K; kk += 32) {
        if (kk + 32 < K) {                            // prefetch next K tile
            __builtin_prefetch(ar0 + kk + 32, 0, 3);
            __builtin_prefetch(wr0 + kk + 32, 0, 3);
        }
        v16bf a0 = load_frag(ar0 + kk, ksel);
        v16bf a1 = load_frag(ar1 + kk, ksel);
        v16bf b0 = load_frag(wr0 + kk, ksel);
        v16bf b1 = load_frag(wr1 + kk, ksel);
        acc00 = WMMA_BF16(a0, b0, acc00);
        acc01 = WMMA_BF16(a0, b1, acc01);
        acc10 = WMMA_BF16(a1, b0, acc10);
        acc11 = WMMA_BF16(a1, b1, acc11);
    }

    int c0 = nt * 32 + lr, c1 = c0 + 16;
    float bias0 = bih[dir * G4H + c0] + bhh[dir * G4H + c0];
    float bias1 = bih[dir * G4H + c1] + bhh[dir * G4H + c1];
    size_t ob = (size_t)dir * Mq * G4H;
#pragma unroll
    for (int r = 0; r < 8; ++r) {
        int row0 = mt * 32 + r + rsel;     // D layout: VGPR r -> M=r (+8 for hi lanes)
        int row1 = row0 + 16;
        out[ob + (size_t)row0 * G4H + c0] = acc00[r] + bias0;
        out[ob + (size_t)row0 * G4H + c1] = acc01[r] + bias1;
        out[ob + (size_t)row1 * G4H + c0] = acc10[r] + bias0;
        out[ob + (size_t)row1 * G4H + c1] = acc11[r] + bias1;
    }
}

// ---------------------------------------------------------------------------
// 3) Recurrent scan: one persistent workgroup per direction (grid=2, 1024 thr).
//    Wave w owns M-half (w&1) and h-slice hb=(w>>1)*16..+15 across ALL FOUR
//    gate chunks (ntiles hb, 16+hb, 32+hb, 48+hb), so i/f/g/o accumulators and
//    the cell state c live entirely in registers. h is shared via 16 KB LDS.
//    The 128 KB xproj row-block for step t+1 is staged into LDS with
//    global_load_async_to_lds_b128 while step t computes (double buffered).
// ---------------------------------------------------------------------------
__global__ __launch_bounds__(1024) void lstm_recurrent(
        const float* __restrict__ xproj,      // [2][4096][1024]
        const __bf16* __restrict__ Whh,       // [2][1024][256]
        __bf16* __restrict__ hout,            // [4096][512] (dir -> col block)
        float* __restrict__ hfinal) {         // [32][512] f32 at t = T-1
    extern __shared__ char smem[];
    float*  xstage[2];
    xstage[0]     = (float*)smem;                         // [32][1024] f32
    xstage[1]     = (float*)(smem + (size_t)Bq * G4H * 4);
    __bf16* h_lds = (__bf16*)(smem + (size_t)2 * Bq * G4H * 4);  // [32][256]

    int dir  = blockIdx.x;
    int tid  = threadIdx.x;
    int wave = tid >> 5, lane = tid & 31;
    int lr = lane & 15, ksel = (lane >> 4) * 8, rsel = (lane >> 4) * 8;

#pragma unroll
    for (int e = 0; e < 8; ++e) h_lds[e * 1024 + tid] = (__bf16)0.0f;

    int mtile = wave & 1;            // which 16-row half of the batch
    int hb    = wave >> 1;           // 16-wide h slice (0..15)
    int hcol  = hb * 16 + lr;
    const float*  xp = xproj + (size_t)dir * Mq * G4H;
    const __bf16* wb = Whh   + (size_t)dir * G4H * Hq;
    const __bf16* wrow[4];
#pragma unroll
    for (int g = 0; g < 4; ++g)      // W row for gate g, this h slice
        wrow[g] = wb + (size_t)(g * 256 + hb * 16 + lr) * Hq;

    float creg[8];
#pragma unroll
    for (int r = 0; r < 8; ++r) creg[r] = 0.0f;

    // stage xproj row-block for t=0
    {
        uint32_t lbase = (uint32_t)(uintptr_t)xstage[0];
#pragma unroll
        for (int i = 0; i < 8; ++i) {
            int off = (tid + i * 1024) * 16;              // 128 KB total per block
            async_copy_b128(lbase + off, (const char*)xp + off);
        }
    }
    wait_asynccnt0();
    __syncthreads();

    for (int t = 0; t < Tq; ++t) {
        const float* xcur = xstage[t & 1];
        if (t + 1 < Tq) {                                 // stage next step's block
            uint32_t lbase = (uint32_t)(uintptr_t)xstage[(t + 1) & 1];
            const char* gsrc = (const char*)(xp + (size_t)(t + 1) * Bq * G4H);
#pragma unroll
            for (int i = 0; i < 8; ++i) {
                int off = (tid + i * 1024) * 16;
                async_copy_b128(lbase + off, gsrc + off);
            }
        }

        // gates = h_prev x Whh^T (WMMA, A from LDS) + staged xproj
        v8f acc[4] = {{}, {}, {}, {}};
        const __bf16* arow = h_lds + (mtile * 16 + lr) * Hq;
#pragma unroll
        for (int kk = 0; kk < Hq; kk += 32) {
            v16bf af = load_frag(arow + kk, ksel);        // ds_load_b128 x2
#pragma unroll
            for (int g = 0; g < 4; ++g) {
                v16bf bf = load_frag(wrow[g] + kk, ksel); // global b128 x2 (L2-hot)
                acc[g] = WMMA_BF16(af, bf, acc[g]);
            }
        }
#pragma unroll
        for (int r = 0; r < 8; ++r) {
            int b = mtile * 16 + r + rsel;
#pragma unroll
            for (int g = 0; g < 4; ++g)
                acc[g][r] += xcur[b * G4H + g * 256 + hcol];
        }
        __syncthreads();   // all h_lds reads done before overwrite

        // gate math fully in registers (i,f,g,o PyTorch order)
#pragma unroll
        for (int r = 0; r < 8; ++r) {
            int b = mtile * 16 + r + rsel;
            float c  = sigmoidf(acc[1][r]) * creg[r]
                     + sigmoidf(acc[0][r]) * tanhf(acc[2][r]);
            float hv = sigmoidf(acc[3][r]) * tanhf(c);
            creg[r] = c;
            h_lds[b * Hq + hcol] = (__bf16)hv;
            hout[((size_t)t * Bq + b) * (2 * Hq) + dir * Hq + hcol] = (__bf16)hv;
            if (t == Tq - 1) hfinal[b * (2 * Hq) + dir * Hq + hcol] = hv;
        }
        wait_asynccnt0();  // next xstage buffer fully landed in LDS
        __syncthreads();
    }
}

// ---------------------------------------------------------------------------
// 4) Heads (tiny, plain f32)
// ---------------------------------------------------------------------------
__global__ void dense_kernel(const float* __restrict__ target,   // [32][768]
                             const float* __restrict__ finalh,   // [32][512]
                             const float* __restrict__ w,        // [640][1280]
                             const float* __restrict__ bias,     // [640]
                             float* __restrict__ feats) {        // [32][640]
    int i = blockIdx.x * blockDim.x + threadIdx.x;               // < 32*640
    int b = i / HALFq, o = i % HALFq;
    const float* wr = w + (size_t)o * Dq;
    const float* tg = target + (size_t)b * Eq;
    const float* fh = finalh + (size_t)b * (2 * Hq);
    float acc = bias[o];
    for (int k = 0; k < Eq; ++k)      acc += tg[k] * wr[k];
    for (int k = 0; k < 2 * Hq; ++k)  acc += fh[k] * wr[Eq + k];
    feats[i] = tanhf(acc);
}

__global__ void clf_kernel(const float* __restrict__ feats,      // [32][640]
                           const float* __restrict__ w,          // [2][640]
                           const float* __restrict__ bias,       // [2]
                           float* __restrict__ out) {            // logits @0, probs @64
    int i = threadIdx.x;
    if (i >= Bq * 2) return;
    int b = i >> 1, j = i & 1;
    const float* f  = feats + b * HALFq;
    const float* wr = w + j * HALFq;
    float acc = bias[j];
    for (int k = 0; k < HALFq; ++k) acc += f[k] * wr[k];
    out[b * 2 + j]          = acc;                               // logits
    out[Bq * 2 + b * 2 + j] = 1.0f / (1.0f + __expf(-acc));      // probs
}

// ---------------------------------------------------------------------------
// Launcher
// ---------------------------------------------------------------------------
extern "C" void kernel_launch(void* const* d_in, const int* in_sizes, int n_in,
                              void* d_out, int out_size, void* d_ws, size_t ws_size,
                              hipStream_t stream) {
    (void)in_sizes; (void)n_in; (void)out_size; (void)ws_size;
    const int*   article   = (const int*)d_in[2];
    const int*   positions = (const int*)d_in[5];
    const float* emb       = (const float*)d_in[8];
    const float* w_ih0     = (const float*)d_in[9];
    const float* w_hh0     = (const float*)d_in[10];
    const float* b_ih0     = (const float*)d_in[11];
    const float* b_hh0     = (const float*)d_in[12];
    const float* w_ih1     = (const float*)d_in[13];
    const float* w_hh1     = (const float*)d_in[14];
    const float* b_ih1     = (const float*)d_in[15];
    const float* b_hh1     = (const float*)d_in[16];
    const float* dense_w   = (const float*)d_in[17];
    const float* dense_b   = (const float*)d_in[18];
    const float* clf_w     = (const float*)d_in[19];
    const float* clf_b     = (const float*)d_in[20];
    float* out = (float*)d_out;

    // Workspace carve-up (256B-aligned chunks)
    uint8_t* ws = (uint8_t*)d_ws;
    size_t off = 0;
    auto carve = [&](size_t bytes) { uint8_t* p = ws + off; off += (bytes + 255) & ~(size_t)255; return p; };
    __bf16* A0bf    = (__bf16*)carve((size_t)Mq * Eq * 2);          // 6.3 MB
    __bf16* A1bf    = (__bf16*)carve((size_t)Mq * 2 * Hq * 2);      // 4.2 MB
    __bf16* Wih0bf  = (__bf16*)carve((size_t)2 * G4H * Eq * 2);     // 3.1 MB
    __bf16* Whh0bf  = (__bf16*)carve((size_t)2 * G4H * Hq * 2);     // 1.0 MB
    __bf16* Wih1bf  = (__bf16*)carve((size_t)2 * G4H * 2 * Hq * 2); // 2.1 MB
    __bf16* Whh1bf  = (__bf16*)carve((size_t)2 * G4H * Hq * 2);     // 1.0 MB
    float*  xproj   = (float*) carve((size_t)2 * Mq * G4H * 4);     // 33.5 MB (reused L0/L1)
    __bf16* hs1bf   = (__bf16*)carve((size_t)Mq * 2 * Hq * 2);      // 4.2 MB
    float*  finalh  = (float*) carve((size_t)Bq * 2 * Hq * 4);
    float*  feats   = (float*) carve((size_t)Bq * HALFq * 4);

    // 1) gather embeddings + target reps (straight into d_out+128)
    gather_embed<<<(Mq * Eq) / 256, 256, 0, stream>>>(article, emb, A0bf);
    target_reps<<<(Bq * Eq) / 256, 256, 0, stream>>>(article, positions, emb, out + 2 * Bq * 2);

    // 2) weight conversions
    cvt_bf16<<<(2 * G4H * Eq) / 256, 256, 0, stream>>>(w_ih0, Wih0bf, 2 * G4H * Eq);
    cvt_bf16<<<(2 * G4H * Hq) / 256, 256, 0, stream>>>(w_hh0, Whh0bf, 2 * G4H * Hq);
    cvt_bf16<<<(2 * G4H * 2 * Hq) / 256, 256, 0, stream>>>(w_ih1, Wih1bf, 2 * G4H * 2 * Hq);
    cvt_bf16<<<(2 * G4H * Hq) / 256, 256, 0, stream>>>(w_hh1, Whh1bf, 2 * G4H * Hq);

    // 3) layer 0: input-projection GEMM + recurrent scan (writes A1bf)
    const int gemm_blocks = (2 * (Mq / 32) * (G4H / 32) * 32) / 256;   // 1024
    gemm_xproj<Eq><<<gemm_blocks, 256, 0, stream>>>(A0bf, Wih0bf, b_ih0, b_hh0, xproj);
    const size_t lds_bytes = (size_t)2 * Bq * G4H * 4 + Bq * Hq * 2;   // 272 KB
    lstm_recurrent<<<2, 1024, lds_bytes, stream>>>(xproj, Whh0bf, A1bf, finalh);

    // 4) layer 1: same pipeline on concat(h_f, h_b)
    gemm_xproj<2 * Hq><<<gemm_blocks, 256, 0, stream>>>(A1bf, Wih1bf, b_ih1, b_hh1, xproj);
    lstm_recurrent<<<2, 1024, lds_bytes, stream>>>(xproj, Whh1bf, hs1bf, finalh);

    // 5) heads
    dense_kernel<<<(Bq * HALFq) / 256, 256, 0, stream>>>(out + 2 * Bq * 2, finalh,
                                                         dense_w, dense_b, feats);
    clf_kernel<<<1, 64, 0, stream>>>(feats, clf_w, clf_b, out);
}
